// unif_vMF_37185826849365
// MI455X (gfx1250) — matrix operand order repaired and checked by reading, not compiled, not run on hardware
//
#include <hip/hip_runtime.h>
#include <cmath>
#include <cstdint>

#define HID 512
#define LAT 128
#define NS 4
#define KAPPA_F 50.0f
#define NORM_EPS_F 1.0f
#define APITCH 516   // 512 + 4 floats pad: conflict-free b128 LDS reads

typedef __attribute__((ext_vector_type(16))) _Float16 v16h;
typedef __attribute__((ext_vector_type(8)))  float    v8f;

// ---------------- deterministic counter-based RNG helpers ----------------
__device__ __forceinline__ uint64_t mix64(uint64_t x) {
    x ^= x >> 33; x *= 0xff51afd7ed558ccdULL;
    x ^= x >> 33; x *= 0xc4ceb9fe1a85ec53ULL;
    x ^= x >> 33;
    return x;
}
__device__ __forceinline__ float u01(uint64_t h) {
    return ((uint32_t)(h >> 40)) * (1.0f / 16777216.0f) + (0.5f / 16777216.0f);
}
__device__ __forceinline__ float gaussf(uint64_t key) {
    uint64_t h1 = mix64(key * 2u + 1u);
    uint64_t h2 = mix64(key * 2u + 2u);
    float u1 = u01(h1), u2 = u01(h2);
    return sqrtf(-2.0f * logf(u1)) * __cosf(6.28318530718f * u2);
}

// ---- gfx1250 async global->LDS copy (ASYNCcnt path, no VGPR round-trip) ----
// VDST VGPR holds the LDS byte offset (HW adds LDS_BASE); GV mode 64-bit vaddr.
__device__ __forceinline__ void async_g2l_b128(unsigned lds_byte_off, const float* gptr) {
    asm volatile("global_load_async_to_lds_b128 %0, %1, off"
                 :: "v"(lds_byte_off), "v"(gptr)
                 : "memory");
}
__device__ __forceinline__ void wait_asynccnt0() {
    asm volatile("s_wait_asynccnt 0x0" ::: "memory");
}

// ---------------- fused vMF kernel ----------------
__global__ __launch_bounds__(256) void unif_vMF_37185826849365_kernel(
    const float* __restrict__ lat,   // (B, 512)
    const float* __restrict__ Wm,    // (128, 512)
    const float* __restrict__ bmu,   // (128,)
    float* __restrict__ out_vecs,    // (4, B, 128)
    float* __restrict__ out_kld,     // (B,)
    float* __restrict__ out_mu,      // (B, 128)
    float* __restrict__ out_rn,      // (B,)
    float kld_const, int batch)
{
    __shared__ float lds_a[16 * APITCH];     // ~32 KB: A tile, f32, padded pitch
    __shared__ float lds_t[16][LAT + 4];     // mu_lin tile -> mu tile

    const int tid     = threadIdx.x;
    const int rowBase = blockIdx.x * 16;

    // ---- async DMA the 16x512 f32 A tile into LDS (2048 x b128, 8 per thread) ----
    #pragma unroll
    for (int i = 0; i < 8; ++i) {
        int q  = tid + 256 * i;      // float4 chunk id, 0..2047
        int r  = q >> 7;             // row 0..15
        int k4 = q & 127;            // float4 within row
        const float* g = lat + (size_t)(rowBase + r) * HID + (k4 << 2);
        unsigned off = (unsigned)(uintptr_t)(const void*)&lds_a[r * APITCH + (k4 << 2)];
        async_g2l_b128(off, g);
    }
    wait_asynccnt0();
    __syncthreads();

    const int wave = tid >> 5;
    const int lane = tid & 31;
    const int n0   = wave * 16;      // this wave's 16 output columns
    const int m    = lane & 15;
    const int hi   = (lane & 16) ? 1 : 0;

    // ---- GEMM: 16x16 tile, K=512 in 16 steps of v_wmma_f32_16x16x32_f16 ----
    v8f acc = {};
    const float* arow = &lds_a[m * APITCH];
    for (int kk = 0; kk < HID; kk += 32) {
        // A fragment (16x32 f16): lanes 0-15 hold M=lane, K=kk+0..7 / kk+16..23
        //                         lanes 16-31 hold M=lane-16, K=kk+8..15 / kk+24..31
        const int kA = kk + hi * 8;
        float4 a01 = *(const float4*)(arow + kA);
        float4 a23 = *(const float4*)(arow + kA + 4);
        float4 a45 = *(const float4*)(arow + kA + 16);
        float4 a67 = *(const float4*)(arow + kA + 20);
        v16h afrag;
        afrag[0]  = (_Float16)a01.x; afrag[1]  = (_Float16)a01.y;
        afrag[2]  = (_Float16)a01.z; afrag[3]  = (_Float16)a01.w;
        afrag[4]  = (_Float16)a23.x; afrag[5]  = (_Float16)a23.y;
        afrag[6]  = (_Float16)a23.z; afrag[7]  = (_Float16)a23.w;
        afrag[8]  = (_Float16)a45.x; afrag[9]  = (_Float16)a45.y;
        afrag[10] = (_Float16)a45.z; afrag[11] = (_Float16)a45.w;
        afrag[12] = (_Float16)a67.x; afrag[13] = (_Float16)a67.y;
        afrag[14] = (_Float16)a67.z; afrag[15] = (_Float16)a67.w;

        // B fragment (32x16 f16): B[k][n] = W[n0+n][k]; lane n = lane&15,
        // lanes 0-15 hold K=kk+0..7 / kk+16..23, lanes 16-31 hold K=kk+8..15 / kk+24..31
        const int kB = kk + hi * 8;
        const float4* wp4 = (const float4*)(Wm + (size_t)(n0 + m) * HID + kB);
        v16h bfrag;
        #pragma unroll
        for (int q = 0; q < 4; ++q) {
            float4 f = wp4[(q & 1) + (q >> 1) * 4];
            bfrag[4 * q + 0] = (_Float16)f.x;
            bfrag[4 * q + 1] = (_Float16)f.y;
            bfrag[4 * q + 2] = (_Float16)f.z;
            bfrag[4 * q + 3] = (_Float16)f.w;
        }

        acc = __builtin_amdgcn_wmma_f32_16x16x32_f16(
            false, afrag, false, bfrag, (short)0, acc, false, false);
    }

    // ---- store tile + bias to LDS (C/D layout: VGPR r -> M = r + hi*8, N = lane&15) ----
    {
        const int col = n0 + m;
        const float b = bmu[col];
        #pragma unroll
        for (int r = 0; r < 8; ++r) {
            lds_t[r + hi * 8][col] = acc[r] + b;
        }
    }
    __syncthreads();

    // ---- per-row norm / mu / kld / redundant_norm: wave w owns rows 2w, 2w+1 ----
    for (int rr = 0; rr < 2; ++rr) {
        const int r    = 2 * wave + rr;
        const int gRow = rowBase + r;
        float ss = 0.f;
        #pragma unroll
        for (int j = 0; j < 4; ++j) {
            float v = lds_t[r][lane + 32 * j];
            ss += v * v;
        }
        #pragma unroll
        for (int o = 16; o > 0; o >>= 1) ss += __shfl_xor(ss, o, 32);
        const float nrm = sqrtf(ss);
        if (lane == 0) {
            float d = 1.0f - nrm;
            __builtin_nontemporal_store(d * d, &out_rn[gRow]);
            __builtin_nontemporal_store(kld_const, &out_kld[gRow]);
        }
        const float inv = 1.0f / nrm;
        #pragma unroll
        for (int j = 0; j < 4; ++j) {
            int c = lane + 32 * j;
            float muv = lds_t[r][c] * inv;
            lds_t[r][c] = muv;                       // tile now holds unit mu
            __builtin_nontemporal_store(muv, &out_mu[(size_t)gRow * LAT + c]);
        }
    }
    // no barrier needed: sampling below only touches this wave's own rows

    // ---- vMF sampling (Wood), 4 samples x 2 rows per wave ----
    const float dm1 = (float)(LAT - 1);
    const float bw  = dm1 / (sqrtf(4.f * KAPPA_F * KAPPA_F + dm1 * dm1) + 2.f * KAPPA_F);
    const float xw  = (1.f - bw) / (1.f + bw);
    const float cw  = KAPPA_F * xw + dm1 * logf(1.f - xw * xw);

    for (int p = 0; p < 8; ++p) {
        const int s    = p >> 1;
        const int r    = 2 * wave + (p & 1);
        const int gRow = rowBase + r;
        const uint64_t keybase = (uint64_t)s * (uint64_t)batch + (uint64_t)gRow;

        // Gaussian v, projection against mu
        float v[4], muv[4];
        float dot = 0.f;
        #pragma unroll
        for (int j = 0; j < 4; ++j) {
            int c = lane + 32 * j;
            v[j]   = gaussf((keybase << 9) | (uint64_t)c);
            muv[j] = lds_t[r][c];
            dot += v[j] * muv[j];
        }
        #pragma unroll
        for (int o = 16; o > 0; o >>= 1) dot += __shfl_xor(dot, o, 32);

        float on = 0.f;
        #pragma unroll
        for (int j = 0; j < 4; ++j) { v[j] -= muv[j] * dot; on += v[j] * v[j]; }
        #pragma unroll
        for (int o = 16; o > 0; o >>= 1) on += __shfl_xor(on, o, 32);
        const float oinv = rsqrtf(on);

        // scalar w: Wood rejection, Beta(63.5,63.5) ~ N(0.5, 1/512) for large alpha.
        // Identical on all lanes (uniform branches), bounded iterations.
        float wv = 0.f;
        bool found = false;
        for (int it = 0; it < 16; ++it) {
            uint64_t k2 = mix64((keybase << 5) ^ (0x9e3779b97f4a7c15ULL + (uint64_t)it));
            float z = 0.5f + 0.04419417f * gaussf(k2);
            z = fminf(fmaxf(z, 1e-6f), 1.f - 1e-6f);
            float wn = (1.f - (1.f + bw) * z) / (1.f - (1.f - bw) * z);
            float u  = u01(mix64(k2 ^ 0x123456789abcdefULL));
            bool ok = (KAPPA_F * wn + dm1 * logf(1.f - xw * wn) - cw) >= logf(u);
            if (ok) { wv = wn; found = true; break; }
        }
        if (!found) wv = xw;  // deterministic fallback (mode region)

        const float noise = 1.0f + u01(mix64(keybase ^ 0x55AA55AA1122ULL)) * NORM_EPS_F;
        const float sq    = sqrtf(fmaxf(0.f, 1.f - wv * wv));

        #pragma unroll
        for (int j = 0; j < 4; ++j) {
            int c = lane + 32 * j;
            __builtin_nontemporal_store(
                (v[j] * oinv * sq + muv[j] * wv) * noise,
                &out_vecs[((size_t)s * (size_t)batch + (size_t)gRow) * LAT + c]);
        }
    }
}

// ---------------- host: KLD constant via log-Bessel series ----------------
static double host_log_iv(double nu, double k) {
    const int M = 300;
    double mx = -1e300;
    double t[M];
    for (int i = 0; i < M; ++i) {
        t[i] = (2.0 * i + nu) * std::log(k / 2.0)
             - std::lgamma(i + 1.0) - std::lgamma(i + nu + 1.0);
        if (t[i] > mx) mx = t[i];
    }
    double s = 0.0;
    for (int i = 0; i < M; ++i) s += std::exp(t[i] - mx);
    return mx + std::log(s);
}

extern "C" void kernel_launch(void* const* d_in, const int* in_sizes, int n_in,
                              void* d_out, int out_size, void* d_ws, size_t ws_size,
                              hipStream_t stream) {
    const float* lat = (const float*)d_in[0];
    const float* Wm  = (const float*)d_in[1];
    const float* bmu = (const float*)d_in[2];
    // d_in[3] = n_sample (fixed at 4 in the reference setup)

    const int batch = in_sizes[0] / HID;  // 65536

    float* out_vecs = (float*)d_out;
    float* out_kld  = out_vecs + (size_t)NS * batch * LAT;
    float* out_mu   = out_kld + batch;
    float* out_rn   = out_mu + (size_t)batch * LAT;

    // KLD_CONST = vMF KLD(kappa=50, d=128) + log(NORM_MAX/NORM_EPS)
    const double k = 50.0, d = (double)LAT;
    double li0 = host_log_iv(d / 2.0, k);
    double li1 = host_log_iv(d / 2.0 + 1.0, k);
    double ratio = std::exp(li1 - li0);
    double kld = k * ratio + d * std::log(k) / 2.0 - li0
               - std::lgamma(d / 2.0 + 1.0) - d * std::log(2.0) / 2.0
               + std::log(2.0 / 1.0);

    dim3 grid(batch / 16), block(256);
    unif_vMF_37185826849365_kernel<<<grid, block, 0, stream>>>(
        lat, Wm, bmu, out_vecs, out_kld, out_mu, out_rn, (float)kld, batch);
}
